// Net_33904471835045
// MI455X (gfx1250) — matrix operand order repaired and checked by reading, not compiled, not run on hardware
//
#include <hip/hip_runtime.h>

// ---------------------------------------------------------------------------
// MI455X (gfx1250, wave32) fully-fused implementation.
//
// reference:  h = relu(x @ W0.T + b0); h = relu(h @ W1.T + b1)
//             o = h @ W2.T + b2                       (B x 99)
//             I[:,0]=0.12 ; I[:,k] = I[:,k-1] + dt*(o[:,k-1]*I*(1-I) - g*I)
//
// Roofline: 3.3 GFLOP vs 105 MB mandatory HBM traffic (x in + I out) ->
// memory bound at 23.3 TB/s (~4.5us). Single fused kernel: WMMA f16-in /
// f32-accum for the MLP, Euler scan done in LDS, one coalesced writeback.
// Weights (45 KB) are re-staged per block but live in the 192 MB L2.
// ---------------------------------------------------------------------------

typedef __attribute__((ext_vector_type(16))) _Float16 v16h;
typedef __attribute__((ext_vector_type(8)))  _Float16 v8h;
typedef __attribute__((ext_vector_type(8)))  float    v8f;

#define B_TOTAL 131072
#define D_IN    100
#define H_DIM   50
#define T_OUT   99      // 99 growth-rate outputs per row

// LDS leading dims: row stride multiple of 16B for aligned v8h loads, and
// half-wave lanes (stride = ld/2 dwords for f16, ld dwords for f32) land in
// distinct LDS bank groups.
#define LDW0 136        // K padded 100 -> 128 (fragment reads reach k=127)
#define LDW1 72         // K padded 50 -> 64
#define LDW2 72
#define LDH  72         // per-wave h tile, 16 x 64 used
#define LDO  113        // per-wave scan tile (f32): 113 mod 64 is odd -> no
                        // bank conflicts among the 16 scanning lanes

__device__ __forceinline__ v16h make16(v8h lo, v8h hi) {
  return __builtin_shufflevector(lo, hi,
      0,1,2,3,4,5,6,7,8,9,10,11,12,13,14,15);
}

// B fragment (32x16, f16): lane holds column N = lane&15,
// K(e) = k0 + (e<8?0:16) + (lane>>4)*8 + (e&7) => two contiguous 8-half runs.
__device__ __forceinline__ v16h load_bfrag(const _Float16* W, int ld, int n, int kstart) {
  const _Float16* p = W + n * ld + (kstart & ~16) + ((kstart & 16) >> 1);
  v8h lo = *(const v8h*)(p);
  v8h hi = *(const v8h*)(p + 16);
  return make16(lo, hi);
}

// A fragment (16x32, f16): lane holds row M = lane&15,
// K(e) = k0 + (e<8?0:16) + hi*8 + (e&7)  => two contiguous 8-half runs.
__device__ __forceinline__ v16h load_afrag(const _Float16* Hb, int ld, int m, int k0, int hi) {
  const _Float16* p = Hb + m * ld + k0 + hi * 8;
  v8h lo  = *(const v8h*)(p);
  v8h hi8 = *(const v8h*)(p + 16);
  return make16(lo, hi8);
}

__device__ __forceinline__ v8f wmma_f16(v16h a, v16h b, v8f c) {
  // (neg_a, A, neg_b, B, c_mod, C, reuse_a, reuse_b)
  return __builtin_amdgcn_wmma_f32_16x16x32_f16(false, a, false, b,
                                                (short)0, c, false, false);
}

__global__ __launch_bounds__(256)
void fused_mlp_euler_kernel(const float* __restrict__ x,
                            const float* __restrict__ W0, const float* __restrict__ b0,
                            const float* __restrict__ W1, const float* __restrict__ b1,
                            const float* __restrict__ W2, const float* __restrict__ b2,
                            float* __restrict__ out)
{
  // Weights + h tiles are dead after the layer-2 WMMAs consume them, so the
  // Euler scan tile is overlaid on the same LDS (union). Peak static LDS:
  // max(61184, 57856) + 960 = 62144 B.
  __shared__ __align__(32) union SMem {
    struct {
      _Float16 w0[64  * LDW0];   // W0 padded 64x136 (f16)
      _Float16 w1[64  * LDW1];   // W1 padded 64x72
      _Float16 w2[112 * LDW2];   // W2 padded 112x72
      _Float16 h[8][16 * LDH];   // per-wave h tile
    } a;
    float o[8][16 * LDO];        // per-wave scan tile (f32)
  } sm;
  __shared__ float sB0[64], sB1[64], sB2[112];   // zero-padded biases

  const int tid = threadIdx.x;

  // ---- stage weights (zero padded: padded K/N lanes contribute exactly 0,
  //      padded bias = 0 so relu(0)=0 keeps padded h columns clean) ----
  for (int i = tid; i < 64 * LDW0; i += 256) {
    int r = i / LDW0, c = i % LDW0;
    sm.a.w0[i] = (_Float16)((r < H_DIM && c < D_IN) ? W0[r * D_IN + c] : 0.f);
  }
  for (int i = tid; i < 64 * LDW1; i += 256) {
    int r = i / LDW1, c = i % LDW1;
    sm.a.w1[i] = (_Float16)((r < H_DIM && c < H_DIM) ? W1[r * H_DIM + c] : 0.f);
  }
  for (int i = tid; i < 112 * LDW2; i += 256) {
    int r = i / LDW2, c = i % LDW2;
    sm.a.w2[i] = (_Float16)((r < T_OUT && c < H_DIM) ? W2[r * H_DIM + c] : 0.f);
  }
  if (tid < 64)  sB0[tid] = (tid < H_DIM) ? b0[tid] : 0.f;
  if (tid < 64)  sB1[tid] = (tid < H_DIM) ? b1[tid] : 0.f;
  if (tid < 112) sB2[tid] = (tid < T_OUT) ? b2[tid] : 0.f;
  __syncthreads();

  const int wave = tid >> 5;
  const int lane = tid & 31;
  const int m    = lane & 15;   // A row / B,C column within the 16-wide tile
  const int hi   = lane >> 4;   // half-wave select
  const int row0 = blockIdx.x * 128 + wave * 16;

  const v8f vzero = {0.f, 0.f, 0.f, 0.f, 0.f, 0.f, 0.f, 0.f};

  // ================= layer 0: h = relu(x @ W0.T + b0), K=100->128 ==========
  v16h a0[4];
#pragma unroll
  for (int s = 0; s < 4; ++s) {
    const float* xr = x + (size_t)(row0 + m) * D_IN;
    v16h t;
#pragma unroll
    for (int e = 0; e < 16; ++e) {
      int v = e >> 1;
      int k = s * 32 + ((v < 4) ? 0 : 16) + hi * 8 + 2 * (v & 3) + (e & 1);
      float xv = (k < D_IN) ? xr[k] : 0.f;
      t[e] = (_Float16)xv;
    }
    a0[s] = t;
  }

  v8f acc0[4];
#pragma unroll
  for (int t4 = 0; t4 < 4; ++t4) {
    acc0[t4] = vzero;
#pragma unroll
    for (int s = 0; s < 4; ++s) {
      v16h bf = load_bfrag(sm.a.w0, LDW0, t4 * 16 + m, s * 32 + hi * 16);
      acc0[t4] = wmma_f16(a0[s], bf, acc0[t4]);
    }
  }
#pragma unroll
  for (int t4 = 0; t4 < 4; ++t4) {
#pragma unroll
    for (int r = 0; r < 8; ++r) {
      int M = r + hi * 8, N = t4 * 16 + m;
      float hv = acc0[t4][r] + sB0[N];
      sm.a.h[wave][M * LDH + N] = (_Float16)fmaxf(hv, 0.f);
    }
  }
  __syncthreads();

  // ================= layer 1: h = relu(h @ W1.T + b1), K=50->64 ============
  v16h a1[2];
#pragma unroll
  for (int s = 0; s < 2; ++s) a1[s] = load_afrag(&sm.a.h[wave][0], LDH, m, s * 32, hi);

  v8f acc1[4];
#pragma unroll
  for (int t4 = 0; t4 < 4; ++t4) {
    acc1[t4] = vzero;
#pragma unroll
    for (int s = 0; s < 2; ++s) {
      v16h bf = load_bfrag(sm.a.w1, LDW1, t4 * 16 + m, s * 32 + hi * 16);
      acc1[t4] = wmma_f16(a1[s], bf, acc1[t4]);
    }
  }
  __syncthreads();
#pragma unroll
  for (int t4 = 0; t4 < 4; ++t4) {
#pragma unroll
    for (int r = 0; r < 8; ++r) {
      int M = r + hi * 8, N = t4 * 16 + m;
      float hv = acc1[t4][r] + sB1[N];
      sm.a.h[wave][M * LDH + N] = (_Float16)fmaxf(hv, 0.f);
    }
  }
  __syncthreads();

  // ================= layer 2: o = h @ W2.T + b2, N=99->112 =================
  v16h a2[2];
#pragma unroll
  for (int s = 0; s < 2; ++s) a2[s] = load_afrag(&sm.a.h[wave][0], LDH, m, s * 32, hi);

  v8f acc2[7];
#pragma unroll
  for (int t7 = 0; t7 < 7; ++t7) {
    acc2[t7] = vzero;
#pragma unroll
    for (int s = 0; s < 2; ++s) {
      v16h bf = load_bfrag(sm.a.w2, LDW2, t7 * 16 + m, s * 32 + hi * 16);
      acc2[t7] = wmma_f16(a2[s], bf, acc2[t7]);
    }
  }

  // REQUIRED barrier: every wave must finish reading w2/h before the union
  // region is repurposed as the scan tile.
  __syncthreads();

#pragma unroll
  for (int t7 = 0; t7 < 7; ++t7) {
#pragma unroll
    for (int r = 0; r < 8; ++r) {
      int M = r + hi * 8, N = t7 * 16 + m;          // N in [0,112); cols >=99 unused
      sm.o[wave][M * LDO + N] = acc2[t7][r] + sB2[N];
    }
  }
  __syncthreads();

  // ================= Euler scan in LDS (lanes 0..15, one row each) =========
  // In place: slot k held o_k, becomes I_{k+1}.
  if (lane < 16) {
    float* rowp = &sm.o[wave][lane * LDO];
    float I = 0.12f;
#pragma unroll 1
    for (int k = 0; k < T_OUT; ++k) {
      float o = rowp[k];
      I = I + 0.01f * (o * I * (1.f - I) - 0.05f * I);
      rowp[k] = I;
    }
  }
  __syncthreads();

  // ================= coalesced writeback of I[16][100] =====================
  for (int idx = lane; idx < 16 * 100; idx += 32) {
    int r = idx / 100, c = idx % 100;
    float v = (c == 0) ? 0.12f : sm.o[wave][r * LDO + (c - 1)];
    out[(size_t)(row0 + r) * 100 + c] = v;
  }
}

extern "C" void kernel_launch(void* const* d_in, const int* in_sizes, int n_in,
                              void* d_out, int out_size, void* d_ws, size_t ws_size,
                              hipStream_t stream) {
  (void)in_sizes; (void)n_in; (void)out_size; (void)d_ws; (void)ws_size;
  const float* x  = (const float*)d_in[0];
  const float* W0 = (const float*)d_in[1];
  const float* b0 = (const float*)d_in[2];
  const float* W1 = (const float*)d_in[3];
  const float* b1 = (const float*)d_in[4];
  const float* W2 = (const float*)d_in[5];
  const float* b2 = (const float*)d_in[6];
  float* out = (float*)d_out;

  fused_mlp_euler_kernel<<<B_TOTAL / 128, 256, 0, stream>>>(
      x, W0, b0, W1, b1, W2, b2, out);
}
